// SAGE_21440476742321
// MI455X (gfx1250) — matrix operand order repaired and checked by reading, not compiled, not run on hardware
//
#include <hip/hip_runtime.h>

#define IN_F   128
#define HID_F  128
#define OUT_F  64

typedef __attribute__((ext_vector_type(16))) _Float16 v16h;
typedef __attribute__((ext_vector_type(8)))  _Float16 v8h;
typedef __attribute__((ext_vector_type(8)))  float    v8f;
typedef __attribute__((ext_vector_type(4))) unsigned int u32x4;
typedef __attribute__((ext_vector_type(8)))  int      i32x8;
typedef __attribute__((ext_vector_type(4)))  int      i32x4;

#if defined(__HIP_DEVICE_COMPILE__) && __has_builtin(__builtin_amdgcn_tensor_load_to_lds)
#define HAVE_TDM 1
#else
#define HAVE_TDM 0
#endif

// ---------------------------------------------------------------------------
// TDM: 2D fp32 tile load (tile_dim0 = 128 elems/row, tile_dim1 = 16 rows)
// Descriptor packing per CDNA5 ISA.
// ---------------------------------------------------------------------------
#if HAVE_TDM
__device__ __forceinline__ void tdm_load_panel(unsigned int lds_off,
                                               const float* gptr,
                                               unsigned int rows_remaining) {
  unsigned long long ga = (unsigned long long)gptr;
  u32x4 g0;
  g0[0] = 1u;                                    // count=1, user descriptor
  g0[1] = lds_off;                               // lds_addr (bytes)
  g0[2] = (unsigned int)(ga & 0xFFFFFFFFull);    // global_addr[31:0]
  g0[3] = (unsigned int)((ga >> 32) & 0x01FFFFFFull) | (2u << 30); // [56:32]|type=2
  i32x8 g1;
  g1[0] = (int)(2u << 16);                       // data_size=4B, mask=0, no flags
  g1[1] = (int)(128u << 16);                     // tensor_dim0[15:0]=128 in [31:16]
  g1[2] = (int)((rows_remaining & 0xFFFFu) << 16);        // dim0 hi=0 | dim1 lo
  g1[3] = (int)(((rows_remaining >> 16) & 0xFFFFu) | (128u << 16)); // dim1 hi | tile_dim0=128
  g1[4] = (int)16u;                              // tile_dim1=16, tile_dim2=0
  g1[5] = (int)128u;                             // tensor_dim0_stride=128
  g1[6] = 0;                                     // stride hi | dim1_stride lo
  g1[7] = 0;
  i32x4 z4 = {0, 0, 0, 0};
#if __clang_major__ >= 23
  i32x8 z8 = {0, 0, 0, 0, 0, 0, 0, 0};
  __builtin_amdgcn_tensor_load_to_lds(g0, g1, z4, z4, z8, 0);
#else
  __builtin_amdgcn_tensor_load_to_lds(g0, g1, z4, z4, 0);
#endif
}
#endif

// B tile from pre-transposed f16 weights WT[N][K]: two 8-elem runs per lane.
// elem e<8 -> K = k0 + half*8 + e ; e>=8 -> K = k0 + 16 + half*8 + (e-8) ; N = j0 + lane%16  (ISA §7.12.2 16-bit B)
__device__ __forceinline__ v16h load_tileB_h(const _Float16* __restrict__ WT,
                                             int ldk, int k0, int j0) {
  const int lane = threadIdx.x & 31;
  const _Float16* p = WT + (size_t)(j0 + (lane & 15)) * ldk + k0 + (lane >> 4) * 8;
  v8h lo = *(const v8h*)p;          // global_load_b128
  v8h hi = *(const v8h*)(p + 16);   // global_load_b128
  return __builtin_shufflevector(lo, hi, 0, 1, 2, 3, 4, 5, 6, 7,
                                 8, 9, 10, 11, 12, 13, 14, 15);
}

// ---------------------------------------------------------------------------
// Utility kernels
// ---------------------------------------------------------------------------
__global__ void zero_kernel(float* __restrict__ p, size_t n) {
  size_t i = (size_t)blockIdx.x * blockDim.x + threadIdx.x;
  size_t stride = (size_t)gridDim.x * blockDim.x;
  for (; i < n; i += stride) p[i] = 0.0f;
}

// W[K][N] fp32 -> WT[N][K] f16 (one-time, weights are constants)
__global__ void convert_wT(const float* __restrict__ W, _Float16* __restrict__ WT,
                           int K, int N) {
  int i = blockIdx.x * blockDim.x + threadIdx.x;
  int stride = gridDim.x * blockDim.x;
  int total = K * N;
  for (; i < total; i += stride) {
    int k = i / N, j = i - k * N;
    WT[(size_t)j * K + k] = (_Float16)W[i];
  }
}

__global__ void degree_kernel(const int* __restrict__ dst, float* __restrict__ deg,
                              int n_edges) {
  int i = blockIdx.x * blockDim.x + threadIdx.x;
  int stride = gridDim.x * blockDim.x;
  for (; i < n_edges; i += stride) atomicAdd(&deg[dst[i]], 1.0f);
}

__global__ void invdeg_kernel(float* __restrict__ deg, int n) {
  int i = blockIdx.x * blockDim.x + threadIdx.x;
  if (i < n) deg[i] = 1.0f / fmaxf(deg[i], 1.0f);
}

// One wave32 per edge; lane q covers features q, q+32, ... (coalesced
// gather + coalesced global_atomic_add_f32 scatter).
__global__ void scatter_add_kernel(const float* __restrict__ feat,
                                   const int* __restrict__ src,
                                   const int* __restrict__ dst,
                                   float* __restrict__ agg,
                                   int n_edges, int F) {
  const int lane    = threadIdx.x & 31;
  const int wave    = (blockIdx.x * (blockDim.x >> 5)) + (threadIdx.x >> 5);
  const int n_waves = gridDim.x * (blockDim.x >> 5);
  for (int e = wave; e < n_edges; e += n_waves) {
    const float* xs = feat + (size_t)src[e] * F;
    float*       ad = agg  + (size_t)dst[e] * F;
    for (int q = lane; q < F; q += 32) atomicAdd(&ad[q], xs[q]);
  }
}

// ---------------------------------------------------------------------------
// Fused SAGE layer:  out = act( selfF @ WsT' + (aggF*inv_deg) @ WnT' + b )
// Per block: 16-row tile. TDM stages fp32 panels into LDS; the block then
// converts them ONCE to f16 (inv_deg folded into the neighbor panel) so the
// hot loop is pure ds_load_b128 -> v_wmma with no VALU conversions.
// LDS layout (bytes): [0,16*KF*4) fp32 self | [.., 32*KF*4) fp32 agg |
//                     [.., +16*KF*2) f16 self | [.., +16*KF*2) f16 agg(scaled)
// ---------------------------------------------------------------------------
template <int KF, int NF, bool RELU>
__global__ void sage_layer(const float* __restrict__ selfF,
                           const float* __restrict__ aggF,
                           const float* __restrict__ invdeg,
                           const _Float16* __restrict__ WsT,
                           const _Float16* __restrict__ WnT,
                           const float* __restrict__ bias,
                           float* __restrict__ out, int n_nodes) {
  extern __shared__ char smem_raw[];
  float*    s32x = (float*)smem_raw;               // fp32 staging (TDM target)
  float*    s32a = s32x + 16 * KF;
  _Float16* h16x = (_Float16*)(s32a + 16 * KF);    // converted panels
  _Float16* h16a = h16x + 16 * KF;

  const int wave = threadIdx.x >> 5;
  const int lane = threadIdx.x & 31;
  const int row0 = blockIdx.x * 16;
  if (row0 >= n_nodes) return;                     // uniform across block
  const int j0 = wave * 16;
  const int m  = lane & 15;
  const int hs = lane >> 4;

#if HAVE_TDM
  if (wave == 0) {
    tdm_load_panel(0u, selfF + (size_t)row0 * KF, (unsigned int)(n_nodes - row0));
    tdm_load_panel((unsigned int)(16 * KF * 4), aggF + (size_t)row0 * KF,
                   (unsigned int)(n_nodes - row0));
    __builtin_amdgcn_s_wait_tensorcnt(0);
  }
  __syncthreads();
  // one-shot cooperative fp32 -> f16 conversion, inv_deg folded in
  for (int i = threadIdx.x; i < 16 * KF; i += blockDim.x) {
    int r = i / KF;
    float idg = invdeg[min(row0 + r, n_nodes - 1)];
    h16x[i] = (_Float16)s32x[i];
    h16a[i] = (_Float16)(s32a[i] * idg);
  }
  __syncthreads();
#else
  for (int i = threadIdx.x; i < 16 * KF; i += blockDim.x) {
    int r = i / KF, c = i - r * KF;
    int rr = min(row0 + r, n_nodes - 1);
    float idg = invdeg[rr];
    h16x[i] = (_Float16)selfF[(size_t)rr * KF + c];
    h16a[i] = (_Float16)(aggF[(size_t)rr * KF + c] * idg);
  }
  __syncthreads();
#endif

  const _Float16* px = h16x + m * KF;
  const _Float16* pa = h16a + m * KF;

  v8f acc = {0.f, 0.f, 0.f, 0.f, 0.f, 0.f, 0.f, 0.f};
#pragma unroll
  for (int k0 = 0; k0 < KF; k0 += 32) {
    // A layout (16-bit A 16x32): elems 0..7 -> K = k0+hs*8+e,
    //                            elems 8..15 -> K = k0+16+hs*8+(e-8)
    const int ka = k0 + hs * 8;
    const int kb = k0 + 16 + hs * 8;
    v8h x0 = *(const v8h*)(px + ka);               // ds_load_b128
    v8h x1 = *(const v8h*)(px + kb);               // ds_load_b128
    v16h aS = __builtin_shufflevector(x0, x1, 0, 1, 2, 3, 4, 5, 6, 7,
                                      8, 9, 10, 11, 12, 13, 14, 15);
    v16h bS = load_tileB_h(WsT, KF, k0, j0);
    acc = __builtin_amdgcn_wmma_f32_16x16x32_f16(false, aS, false, bS,
                                                 (short)0, acc, false, false);
    v8h a0 = *(const v8h*)(pa + ka);               // ds_load_b128
    v8h a1 = *(const v8h*)(pa + kb);               // ds_load_b128
    v16h aN = __builtin_shufflevector(a0, a1, 0, 1, 2, 3, 4, 5, 6, 7,
                                      8, 9, 10, 11, 12, 13, 14, 15);
    v16h bN = load_tileB_h(WnT, KF, k0, j0);
    acc = __builtin_amdgcn_wmma_f32_16x16x32_f16(false, aN, false, bN,
                                                 (short)0, acc, false, false);
  }

  const float bj = bias[j0 + m];
#pragma unroll
  for (int r = 0; r < 8; ++r) {
    const int row = row0 + r + hs * 8;             // C/D: VGPR r -> M = r + 8*half
    if (row < n_nodes) {
      float v = acc[r] + bj;
      out[(size_t)row * NF + j0 + m] = RELU ? fmaxf(v, 0.0f) : v;
    }
  }
}

// ---------------------------------------------------------------------------
extern "C" void kernel_launch(void* const* d_in, const int* in_sizes, int n_in,
                              void* d_out, int out_size, void* d_ws, size_t ws_size,
                              hipStream_t stream) {
  (void)n_in; (void)out_size; (void)ws_size;
  const float* x   = (const float*)d_in[0];
  const int*   src = (const int*)  d_in[1];
  const int*   dst = (const int*)  d_in[2];
  const float* W1s = (const float*)d_in[3];
  const float* W1n = (const float*)d_in[4];
  const float* b1  = (const float*)d_in[5];
  const float* W2s = (const float*)d_in[6];
  const float* W2n = (const float*)d_in[7];
  const float* b2  = (const float*)d_in[8];
  float* out = (float*)d_out;

  const int n_nodes = in_sizes[0] / IN_F;
  const int n_edges = in_sizes[1];

  // Workspace: [deg : n][agg : n*128][h : n*128][f16 transposed weights]
  float* deg = (float*)d_ws;
  float* agg = deg + n_nodes;
  float* h   = agg + (size_t)n_nodes * HID_F;
  _Float16* w1sT = (_Float16*)(h + (size_t)n_nodes * HID_F);
  _Float16* w1nT = w1sT + IN_F * HID_F;
  _Float16* w2sT = w1nT + IN_F * HID_F;
  _Float16* w2nT = w2sT + HID_F * OUT_F;

  // LDS per block: fp32 staging (16KB) + f16 panels (8KB)
  const int lds_bytes = 2 * 16 * 128 * 4 + 2 * 16 * 128 * 2;

  // 0) one-time weight convert+transpose (fp32 [K][N] -> f16 [N][K])
  convert_wT<<<64, 256, 0, stream>>>(W1s, w1sT, IN_F, HID_F);
  convert_wT<<<64, 256, 0, stream>>>(W1n, w1nT, IN_F, HID_F);
  convert_wT<<<64, 256, 0, stream>>>(W2s, w2sT, HID_F, OUT_F);
  convert_wT<<<64, 256, 0, stream>>>(W2n, w2nT, HID_F, OUT_F);

  // 1) zero deg + agg (contiguous), accumulate degrees, invert in place
  zero_kernel<<<2048, 256, 0, stream>>>(deg, (size_t)n_nodes * (1 + HID_F));
  degree_kernel<<<2048, 256, 0, stream>>>(dst, deg, n_edges);
  invdeg_kernel<<<(n_nodes + 255) / 256, 256, 0, stream>>>(deg, n_nodes);

  // 2) layer 1: scatter-mean of x, fused dual-WMMA + bias + relu (128->128)
  scatter_add_kernel<<<4096, 256, 0, stream>>>(x, src, dst, agg, n_edges, IN_F);
  sage_layer<IN_F, HID_F, true>
      <<<(n_nodes + 15) / 16, 256, lds_bytes, stream>>>(
          x, agg, deg, w1sT, w1nT, b1, h, n_nodes);

  // 3) layer 2: re-zero agg, scatter-mean of h, dual-WMMA + bias (128->64)
  zero_kernel<<<2048, 256, 0, stream>>>(agg, (size_t)n_nodes * HID_F);
  scatter_add_kernel<<<4096, 256, 0, stream>>>(h, src, dst, agg, n_edges, HID_F);
  sage_layer<HID_F, OUT_F, false>
      <<<(n_nodes + 15) / 16, 128, lds_bytes, stream>>>(
          h, agg, deg, w2sT, w2nT, b2, out, n_nodes);
}
